// Decoder_17162689315573
// MI455X (gfx1250) — matrix operand order repaired and checked by reading, not compiled, not run on hardware
//
#include <hip/hip_runtime.h>
#include <hip/hip_bf16.h>

#define B_  256
#define S_  100
#define E_  512
#define H_  512
#define T_  100
#define FH_ 2048   // 4*H

typedef __bf16 v16bf __attribute__((ext_vector_type(16)));
typedef float  v8f   __attribute__((ext_vector_type(8)));
typedef unsigned int uintx4 __attribute__((ext_vector_type(4)));

union ABV { v16bf v; uintx4 q[2]; };

// Native gfx1250 tanh (single TRANS instruction) if the toolchain exposes it.
__device__ __forceinline__ float fast_tanh(float x) {
#if __has_builtin(__builtin_amdgcn_tanhf)
  return __builtin_amdgcn_tanhf(x);
#else
  return tanhf(x);
#endif
}

// ---------------------------------------------------------------------------
// WMMA fragment loaders.  A row-major (MxK), Bt row-major (NxK).
// Per ISA layout: lane r=L&15 selects row; half=L>>4 selects K sub-ranges.
//   A frag (lane):  K = {ka..ka+7, ka+16..ka+23},  ka = k + half*8
//   B frag (lane):  K = {kb..kb+7, kb+16..kb+23},  kb = k + half*8
// ---------------------------------------------------------------------------
__device__ __forceinline__ v16bf load_afrag(const __bf16* Arow, int k, int half) {
  ABV a;
  int ka = k + half * 8;
  a.q[0] = *(const uintx4*)(Arow + ka);
  a.q[1] = *(const uintx4*)(Arow + ka + 16);
  return a.v;
}
__device__ __forceinline__ v16bf load_bfrag(const __bf16* Brow, int k, int half) {
  ABV b;
  int kb = k + half * 8;
  b.q[0] = *(const uintx4*)(Brow + kb);
  b.q[1] = *(const uintx4*)(Brow + kb + 16);
  return b.v;
}
__device__ __forceinline__ v8f wmma(v16bf a, v16bf b, v8f c) {
  return __builtin_amdgcn_wmma_f32_16x16x32_bf16(false, a, false, b, (short)0, c,
                                                 false, false);
}

// ---------------------------------------------------------------------------
// Generic GEMM:  C(MxN, f32) = A(MxK, bf16) x Bt(NxK, bf16)^T + bias(N)
// block = 128 threads = 4 waves; each wave computes a 16x64 strip:
// 4 WMMAs per K-step sharing one A fragment (4x A reuse in registers).
// N must be a multiple of 256, M of 16, K of 32.
// ---------------------------------------------------------------------------
__global__ void gemm_bias_kernel(const __bf16* __restrict__ A,
                                 const __bf16* __restrict__ Bt,
                                 const float* __restrict__ bias,
                                 float* __restrict__ C,
                                 int M, int N, int K) {
  int wave = threadIdx.x >> 5;
  int lane = threadIdx.x & 31;
  int half = lane >> 4;
  int r    = lane & 15;
  int m0 = blockIdx.y * 16;
  int n0 = blockIdx.x * 256 + wave * 64;
  const __bf16* Arow = A  + (size_t)(m0 + r) * K;
  const __bf16* Brow = Bt + (size_t)(n0 + r) * K;
  v8f acc[4];
#pragma unroll
  for (int j = 0; j < 4; ++j) acc[j] = (v8f){0.f,0.f,0.f,0.f,0.f,0.f,0.f,0.f};
  for (int k = 0; k < K; k += 32) {
    v16bf a = load_afrag(Arow, k, half);
#pragma unroll
    for (int j = 0; j < 4; ++j) {
      v16bf b = load_bfrag(Brow + (size_t)(16 * j) * K, k, half);
      acc[j] = wmma(a, b, acc[j]);
    }
  }
#pragma unroll
  for (int j = 0; j < 4; ++j) {
    int col = n0 + 16 * j + r;
    float bv = bias[col];
#pragma unroll
    for (int v = 0; v < 8; ++v) {
      int row = m0 + half * 8 + v;
      C[(size_t)row * N + col] = acc[j][v] + bv;
    }
  }
}

// LSTM gate GEMM: gates = X@WinT^T + Hx@WhT^T + b_in + b_h  (M=B_, N=FH_, K=512)
__global__ void gemm_gates_kernel(const __bf16* __restrict__ X,
                                  const __bf16* __restrict__ Hx,
                                  const __bf16* __restrict__ WinT,
                                  const __bf16* __restrict__ WhT,
                                  const float* __restrict__ b_in,
                                  const float* __restrict__ b_h,
                                  float* __restrict__ gates) {
  int wave = threadIdx.x >> 5;
  int lane = threadIdx.x & 31;
  int half = lane >> 4;
  int r    = lane & 15;
  int m0 = blockIdx.y * 16;
  int n0 = blockIdx.x * 256 + wave * 64;
  const __bf16* Xrow = X    + (size_t)(m0 + r) * E_;
  const __bf16* Hrow = Hx   + (size_t)(m0 + r) * H_;
  const __bf16* Wi   = WinT + (size_t)(n0 + r) * E_;
  const __bf16* Wh   = WhT  + (size_t)(n0 + r) * H_;
  v8f acc[4];
#pragma unroll
  for (int j = 0; j < 4; ++j) acc[j] = (v8f){0.f,0.f,0.f,0.f,0.f,0.f,0.f,0.f};
  for (int k = 0; k < E_; k += 32) {
    v16bf a = load_afrag(Xrow, k, half);
#pragma unroll
    for (int j = 0; j < 4; ++j)
      acc[j] = wmma(a, load_bfrag(Wi + (size_t)(16 * j) * E_, k, half), acc[j]);
  }
  for (int k = 0; k < H_; k += 32) {
    v16bf a = load_afrag(Hrow, k, half);
#pragma unroll
    for (int j = 0; j < 4; ++j)
      acc[j] = wmma(a, load_bfrag(Wh + (size_t)(16 * j) * H_, k, half), acc[j]);
  }
#pragma unroll
  for (int j = 0; j < 4; ++j) {
    int col = n0 + 16 * j + r;
    float bv = b_in[col] + b_h[col];
#pragma unroll
    for (int v = 0; v < 8; ++v) {
      int row = m0 + half * 8 + v;
      gates[(size_t)row * FH_ + col] = acc[j][v] + bv;
    }
  }
}

// ---------------------------------------------------------------------------
// Elementwise LSTM cell
// ---------------------------------------------------------------------------
__global__ void lstm_kernel(const float* __restrict__ gates,
                            float* __restrict__ cxs,
                            float* __restrict__ hyf,
                            __bf16* __restrict__ hyb) {
  int i = blockIdx.x * 256 + threadIdx.x;
  if (i >= B_ * H_) return;
  int b = i / H_, h = i - b * H_;
  const float* g = gates + (size_t)b * FH_;
  float ig = g[h], fg = g[H_ + h], cg = g[2 * H_ + h], og = g[3 * H_ + h];
  float si = 1.f / (1.f + __expf(-ig));
  float sf = 1.f / (1.f + __expf(-fg));
  float so = 1.f / (1.f + __expf(-og));
  float cy = sf * cxs[i] + si * fast_tanh(cg);
  float hy = so * fast_tanh(cy);
  cxs[i] = cy;
  hyf[i] = hy;
  hyb[i] = (__bf16)hy;
}

// ---------------------------------------------------------------------------
// Glimpse attention: u[s]=sum_h v[h]*tanh(q[b,h]+e[s,b,h]); softmax; gl=e@p
// one block (256 thr = 8 waves) per batch element
// ---------------------------------------------------------------------------
__global__ void attn_g_kernel(const float* __restrict__ q,
                              const float* __restrict__ e,
                              const float* __restrict__ v,
                              const unsigned char* __restrict__ mask,
                              __bf16* __restrict__ gl_bf) {
  int b = blockIdx.x;
  int tid = threadIdx.x, wave = tid >> 5, lane = tid & 31;
  __shared__ float u[S_];
  __shared__ float p[S_];
  const float* qr = q + (size_t)b * H_;
  for (int s = wave; s < S_; s += 8) {
    const float* er = e + ((size_t)s * B_ + b) * H_;
    float sum = 0.f;
    for (int h = lane; h < H_; h += 32) sum += v[h] * fast_tanh(qr[h] + er[h]);
#pragma unroll
    for (int o = 16; o > 0; o >>= 1) sum += __shfl_xor(sum, o, 32);
    if (lane == 0) u[s] = sum;
  }
  __syncthreads();
  if (wave == 0) {
    float mx = -1e30f;
    for (int s = lane; s < S_; s += 32) {
      float val = mask[b * S_ + s] ? -1e30f : u[s];
      u[s] = val;
      mx = fmaxf(mx, val);
    }
#pragma unroll
    for (int o = 16; o > 0; o >>= 1) mx = fmaxf(mx, __shfl_xor(mx, o, 32));
    float sm = 0.f;
    for (int s = lane; s < S_; s += 32) { float ex = __expf(u[s] - mx); p[s] = ex; sm += ex; }
#pragma unroll
    for (int o = 16; o > 0; o >>= 1) sm += __shfl_xor(sm, o, 32);
    float inv = 1.f / sm;
    for (int s = lane; s < S_; s += 32) p[s] *= inv;
  }
  __syncthreads();
  for (int h = tid; h < H_; h += 256) {
    float g = 0.f;
    for (int s = 0; s < S_; ++s) g += e[((size_t)s * B_ + b) * H_ + h] * p[s];
    gl_bf[(size_t)b * H_ + h] = (__bf16)g;
  }
}

__device__ __forceinline__ unsigned wang_hash(unsigned x) {
  x = (x ^ 61u) ^ (x >> 16); x *= 9u; x ^= x >> 4; x *= 0x27d4eb2du; x ^= x >> 15;
  return x;
}

// ---------------------------------------------------------------------------
// Pointer attention + softmax + Gumbel-max sample + mask update + gather
// ---------------------------------------------------------------------------
__global__ void attn_p_sample_kernel(const float* __restrict__ q,
                                     const float* __restrict__ e,
                                     const float* __restrict__ v,
                                     unsigned char* __restrict__ mask,
                                     const float* __restrict__ emb,
                                     __bf16* __restrict__ x_bf,
                                     float* __restrict__ probs_out,
                                     float* __restrict__ sel_out,
                                     int step) {
  int b = blockIdx.x;
  int tid = threadIdx.x, wave = tid >> 5, lane = tid & 31;
  __shared__ float u[S_];
  __shared__ float p[S_];
  __shared__ int selIdx;
  const float* qr = q + (size_t)b * H_;
  for (int s = wave; s < S_; s += 8) {
    const float* er = e + ((size_t)s * B_ + b) * H_;
    float sum = 0.f;
    for (int h = lane; h < H_; h += 32) sum += v[h] * fast_tanh(qr[h] + er[h]);
#pragma unroll
    for (int o = 16; o > 0; o >>= 1) sum += __shfl_xor(sum, o, 32);
    if (lane == 0) u[s] = sum;
  }
  __syncthreads();
  if (wave == 0) {
    float mx = -1e30f;
    for (int s = lane; s < S_; s += 32) {
      float lg = mask[b * S_ + s] ? -1e30f : 10.0f * fast_tanh(u[s]);
      u[s] = lg;
      mx = fmaxf(mx, lg);
    }
#pragma unroll
    for (int o = 16; o > 0; o >>= 1) mx = fmaxf(mx, __shfl_xor(mx, o, 32));
    float sm = 0.f;
    for (int s = lane; s < S_; s += 32) { float ex = __expf(u[s] - mx); p[s] = ex; sm += ex; }
#pragma unroll
    for (int o = 16; o > 0; o >>= 1) sm += __shfl_xor(sm, o, 32);
    float inv = 1.f / sm;
    float best = -1e30f; int bestI = 0;
    for (int s = lane; s < S_; s += 32) {
      p[s] *= inv;
      unsigned h = wang_hash((unsigned)(step * 1664525) ^
                             (unsigned)(b * 1013904223) ^
                             (unsigned)(s * 374761393));
      float uf = ((float)h + 0.5f) * (1.0f / 4294967296.0f);
      float gk = u[s] - logf(-logf(uf));
      if (gk > best) { best = gk; bestI = s; }
    }
#pragma unroll
    for (int o = 16; o > 0; o >>= 1) {
      float ob = __shfl_xor(best, o, 32);
      int   oi = __shfl_xor(bestI, o, 32);
      if (ob > best) { best = ob; bestI = oi; }
    }
    if (lane == 0) {
      selIdx = bestI;
      sel_out[(size_t)step * B_ + b] = (float)bestI;
      mask[b * S_ + bestI] = 1;
    }
  }
  __syncthreads();
  for (int s = tid; s < S_; s += 256)
    probs_out[((size_t)step * B_ + b) * S_ + s] = p[s];
  int idx = selIdx;
  for (int el = tid; el < E_; el += 256)
    x_bf[(size_t)b * E_ + el] = (__bf16)emb[((size_t)idx * B_ + b) * E_ + el];
}

// ---------------------------------------------------------------------------
// One-time conversion kernels
// ---------------------------------------------------------------------------
__global__ void transpose_convert_kernel(const float* __restrict__ in,
                                         __bf16* __restrict__ out, int K, int N) {
  // in: K x N (f32) -> out: N x K (bf16)
  int idx = blockIdx.x * 256 + threadIdx.x;
  if (idx >= K * N) return;
  int n = idx / K, k = idx - n * K;
  out[idx] = (__bf16)in[(size_t)k * N + n];
}

__global__ void convert_flat_kernel(const float* __restrict__ in,
                                    __bf16* __restrict__ out, int n) {
  int i = blockIdx.x * 256 + threadIdx.x;
  if (i < n) out[i] = (__bf16)in[i];
}

__global__ void init_kernel(const float* __restrict__ dec_in,
                            const float* __restrict__ hx,
                            const float* __restrict__ cx,
                            __bf16* __restrict__ x_bf,
                            __bf16* __restrict__ hy_bf,
                            float* __restrict__ hyf,
                            float* __restrict__ cxs,
                            unsigned char* __restrict__ mask) {
  int i = blockIdx.x * 256 + threadIdx.x;
  if (i < B_ * E_) x_bf[i] = (__bf16)dec_in[i];
  if (i < B_ * H_) { hy_bf[i] = (__bf16)hx[i]; hyf[i] = hx[i]; cxs[i] = cx[i]; }
  if (i < B_ * S_) mask[i] = 0;
}

__global__ void final_copy_kernel(const float* __restrict__ hyf,
                                  const float* __restrict__ cxs,
                                  float* __restrict__ oh, float* __restrict__ oc) {
  int i = blockIdx.x * 256 + threadIdx.x;
  if (i < B_ * H_) { oh[i] = hyf[i]; oc[i] = cxs[i]; }
}

// ---------------------------------------------------------------------------
extern "C" void kernel_launch(void* const* d_in, const int* in_sizes, int n_in,
                              void* d_out, int out_size, void* d_ws, size_t ws_size,
                              hipStream_t stream) {
  const float* dec_in = (const float*)d_in[0];
  const float* emb    = (const float*)d_in[1];
  const float* hx     = (const float*)d_in[2];
  const float* cx     = (const float*)d_in[3];
  const float* ctx    = (const float*)d_in[4];
  const float* W_in   = (const float*)d_in[5];
  const float* b_in   = (const float*)d_in[6];
  const float* W_h    = (const float*)d_in[7];
  const float* b_h    = (const float*)d_in[8];
  const float* Wq_g = (const float*)d_in[9];
  const float* bq_g = (const float*)d_in[10];
  const float* Wr_g = (const float*)d_in[11];
  const float* br_g = (const float*)d_in[12];
  const float* v_g  = (const float*)d_in[13];
  const float* Wq_p = (const float*)d_in[14];
  const float* bq_p = (const float*)d_in[15];
  const float* Wr_p = (const float*)d_in[16];
  const float* br_p = (const float*)d_in[17];
  const float* v_p  = (const float*)d_in[18];

  char* ws = (char*)d_ws;
  size_t off = 0;
  auto alloc = [&](size_t bytes) -> void* {
    void* ptr = ws + off;
    off = (off + bytes + 255) & ~(size_t)255;
    return ptr;
  };
  const size_t SBH = (size_t)S_ * B_ * H_;
  float*  e_g    = (float*) alloc(SBH * 4);
  float*  e_p    = (float*) alloc(SBH * 4);
  __bf16* ctx_bf = (__bf16*)alloc(SBH * 2);
  __bf16* WinT   = (__bf16*)alloc((size_t)FH_ * E_ * 2);
  __bf16* WhT    = (__bf16*)alloc((size_t)FH_ * H_ * 2);
  __bf16* WqgT   = (__bf16*)alloc((size_t)H_ * H_ * 2);
  __bf16* WqpT   = (__bf16*)alloc((size_t)H_ * H_ * 2);
  __bf16* Wrg_bf = (__bf16*)alloc((size_t)H_ * H_ * 2);
  __bf16* Wrp_bf = (__bf16*)alloc((size_t)H_ * H_ * 2);
  float*  gates  = (float*) alloc((size_t)B_ * FH_ * 4);
  float*  cxs    = (float*) alloc((size_t)B_ * H_ * 4);
  float*  hyf    = (float*) alloc((size_t)B_ * H_ * 4);
  __bf16* hy_bf  = (__bf16*)alloc((size_t)B_ * H_ * 2);
  __bf16* x_bf   = (__bf16*)alloc((size_t)B_ * E_ * 2);
  float*  qbuf   = (float*) alloc((size_t)B_ * H_ * 4);
  __bf16* gl_bf  = (__bf16*)alloc((size_t)B_ * H_ * 2);
  unsigned char* mask = (unsigned char*)alloc((size_t)B_ * S_);

  float* probs_out = (float*)d_out;                        // (T,B,S)
  float* sel_out   = probs_out + (size_t)T_ * B_ * S_;     // (T,B)
  float* hx_out    = sel_out + (size_t)T_ * B_;            // (B,H)
  float* cx_out    = hx_out + (size_t)B_ * H_;             // (B,H)

  dim3 blk(128);

  // One-time weight conversion / transposition (bf16, N x K layout)
  int nWin = FH_ * E_;
  transpose_convert_kernel<<<(nWin + 255) / 256, 256, 0, stream>>>(W_in, WinT, E_, FH_);
  transpose_convert_kernel<<<(nWin + 255) / 256, 256, 0, stream>>>(W_h,  WhT,  H_, FH_);
  int nWq = H_ * H_;
  transpose_convert_kernel<<<(nWq + 255) / 256, 256, 0, stream>>>(Wq_g, WqgT, H_, H_);
  transpose_convert_kernel<<<(nWq + 255) / 256, 256, 0, stream>>>(Wq_p, WqpT, H_, H_);
  convert_flat_kernel<<<(nWq + 255) / 256, 256, 0, stream>>>(Wr_g, Wrg_bf, nWq);
  convert_flat_kernel<<<(nWq + 255) / 256, 256, 0, stream>>>(Wr_p, Wrp_bf, nWq);
  int nCtx = (int)SBH;
  convert_flat_kernel<<<(nCtx + 255) / 256, 256, 0, stream>>>(ctx, ctx_bf, nCtx);
  init_kernel<<<(B_ * E_ + 255) / 256, 256, 0, stream>>>(dec_in, hx, cx, x_bf, hy_bf,
                                                         hyf, cxs, mask);

  // Hoisted: e = context @ Wr^T + br  (two 25600x512x512 WMMA GEMMs; result
  // 105 MB fp32 stays resident in the 192 MB L2 for the whole decode loop)
  gemm_bias_kernel<<<dim3(H_ / 256, (S_ * B_) / 16), blk, 0, stream>>>(
      ctx_bf, Wrg_bf, br_g, e_g, S_ * B_, H_, H_);
  gemm_bias_kernel<<<dim3(H_ / 256, (S_ * B_) / 16), blk, 0, stream>>>(
      ctx_bf, Wrp_bf, br_p, e_p, S_ * B_, H_, H_);

  for (int t = 0; t < T_; ++t) {
    gemm_gates_kernel<<<dim3(FH_ / 256, B_ / 16), blk, 0, stream>>>(
        x_bf, hy_bf, WinT, WhT, b_in, b_h, gates);
    lstm_kernel<<<(B_ * H_ + 255) / 256, 256, 0, stream>>>(gates, cxs, hyf, hy_bf);
    gemm_bias_kernel<<<dim3(H_ / 256, B_ / 16), blk, 0, stream>>>(
        hy_bf, WqgT, bq_g, qbuf, B_, H_, H_);
    attn_g_kernel<<<B_, 256, 0, stream>>>(qbuf, e_g, v_g, mask, gl_bf);
    gemm_bias_kernel<<<dim3(H_ / 256, B_ / 16), blk, 0, stream>>>(
        gl_bf, WqpT, bq_p, qbuf, B_, H_, H_);
    attn_p_sample_kernel<<<B_, 256, 0, stream>>>(qbuf, e_p, v_p, mask, emb, x_bf,
                                                 probs_out, sel_out, t);
  }
  final_copy_kernel<<<(B_ * H_ + 255) / 256, 256, 0, stream>>>(hyf, cxs, hx_out, cx_out);
}
